// SparseMLP_57956288692247
// MI455X (gfx1250) — matrix operand order repaired and checked by reading, not compiled, not run on hardware
//
#include <hip/hip_runtime.h>

typedef __attribute__((ext_vector_type(16))) _Float16 v16h;
typedef __attribute__((ext_vector_type(8)))  _Float16 v8h;
typedef __attribute__((ext_vector_type(4)))  _Float16 v4h;
typedef __attribute__((ext_vector_type(8)))  float    v8f;

#define IN_F   1536
#define HID    640
#define OUT_F  10
#define BATCH  65536
#define MTILE  64
#define KC     128
#define NCHUNK (IN_F / KC)                   // 12

#define W1_BYTES (HID * IN_F * 2)            // 1,966,080
#define W2_BYTES (HID * HID * 2)             //   819,200
#define W3_ROWS  16                          // pad 10 -> 16

// LDS layout (dynamic):
#define XF32_BYTES (MTILE * KC * 4)          // 32 KB per f32 stage buffer
#define OFF_XF32   0                         // [2][64][128] f32  (64 KB)
#define OFF_XS     (2 * XF32_BYTES)          // [64][128] f16     (16 KB)
#define OFF_H1     (OFF_XS + MTILE * KC * 2) // [64][640] f16     (80 KB)
#define OFF_H2     (OFF_H1 + MTILE * HID * 2)
#define LDS_BYTES  (OFF_H2 + MTILE * HID * 2)   // 245,760 B

union HFrag { v16h v; v8h h[2]; };

__device__ __forceinline__ float gelu_exact(float v) {
  return 0.5f * v * (1.0f + erff(v * 0.70710678118654752440f));
}

// Async DMA of one x chunk (64 rows x KC f32) into LDS, 8 x b128 per thread.
__device__ __forceinline__ void stage_chunk_async(const float* __restrict__ x,
                                                  int m0, int kc,
                                                  char* ldsbuf, int tid) {
  #pragma unroll
  for (int i = 0; i < 8; ++i) {
    int f   = tid + i * 256;       // 0..2047 float4 slots
    int row = f >> 5;              // 32 float4 per row
    int col = (f & 31) * 4;
    const float* g = x + (size_t)(m0 + row) * IN_F + kc + col;
    unsigned dst = (unsigned)(size_t)(ldsbuf + (size_t)f * 16);
    asm volatile("global_load_async_to_lds_b128 %0, %1, off"
                 :: "v"(dst), "v"(g) : "memory");
  }
}

// One GEMM pass over NK k-steps of 32, 4 M-subtiles x 5 N-tiles per wave.
// aBase = A-lds  + l16*LDA + hsel*8    (per-lane; +ms*16*LDA, +ks inside)
// bBase = B-glob + (wave*80 + l16)*LDB + hsel*8    (per-lane; +t*16*LDB, +ks inside)
// B fragments are triple-buffered with prefetch distance 2 to hide L2 latency.
template <int NK, int LDA, int LDB>
__device__ __forceinline__ void gemm_block(const _Float16* __restrict__ aBase,
                                           const _Float16* __restrict__ bBase,
                                           v8f acc[4][5]) {
  constexpr int N5 = NK * 5;
  HFrag bfr[3];
  HFrag afr[4];
  // Prologue: prefetch B for j=0 and j=1.
  bfr[0].h[0] = *(const v8h*)(bBase);
  bfr[0].h[1] = *(const v8h*)(bBase + 16);
  {
    const _Float16* bb = bBase + 1 * 16 * LDB;   // j=1 -> ks=0, t=1
    bfr[1].h[0] = *(const v8h*)(bb);
    bfr[1].h[1] = *(const v8h*)(bb + 16);
  }
  #pragma unroll
  for (int j = 0; j < N5; ++j) {
    const int ks = (j / 5) * 32;
    const int t  = j % 5;
    if (t == 0) {
      #pragma unroll
      for (int ms = 0; ms < 4; ++ms) {
        const _Float16* ab = aBase + ms * 16 * LDA + ks;
        afr[ms].h[0] = *(const v8h*)(ab);        // K: ks + 8*hsel .. +7
        afr[ms].h[1] = *(const v8h*)(ab + 16);   // K: +16
      }
    }
    if (j + 2 < N5) {
      const int ks2 = ((j + 2) / 5) * 32;
      const int t2  = (j + 2) % 5;
      const _Float16* bb = bBase + t2 * 16 * LDB + ks2;
      bfr[(j + 2) % 3].h[0] = *(const v8h*)(bb);
      bfr[(j + 2) % 3].h[1] = *(const v8h*)(bb + 16);
    }
    #pragma unroll
    for (int ms = 0; ms < 4; ++ms)
      acc[ms][t] = __builtin_amdgcn_wmma_f32_16x16x32_f16(
          false, afr[ms].v, false, bfr[j % 3].v, (short)0, acc[ms][t], false, false);
  }
}

// Build masked fp16 weights (row-major, K contiguous), zero-padded rows for W3.
__global__ void sparse_mlp_prep(const float* __restrict__ W,
                                const unsigned char* __restrict__ mask,
                                _Float16* __restrict__ dst,
                                int cols, int padRows, int realRows) {
  int idx = blockIdx.x * blockDim.x + threadIdx.x;
  if (idx >= padRows * cols) return;
  int r = idx / cols;
  float v = 0.0f;
  if (r < realRows && mask[idx] != 0) v = W[idx];
  dst[idx] = (_Float16)v;
}

__global__ __launch_bounds__(256, 1)
void sparse_mlp_fused(const float* __restrict__ x,
                      const _Float16* __restrict__ w1,
                      const _Float16* __restrict__ w2,
                      const _Float16* __restrict__ w3,
                      float* __restrict__ out) {
  extern __shared__ char lds[];
  _Float16* xs = (_Float16*)(lds + OFF_XS);     // [64][KC] f16
  _Float16* h1 = (_Float16*)(lds + OFF_H1);     // [64][HID]
  _Float16* h2 = (_Float16*)(lds + OFF_H2);     // [64][HID]

  const int tid  = threadIdx.x;
  const int lane = tid & 31;
  const int wave = tid >> 5;
  const int hsel = lane >> 4;    // 0: lanes 0-15, 1: lanes 16-31
  const int l16  = lane & 15;
  const int m0   = blockIdx.x * MTILE;

  v8f acc[4][5];
  #pragma unroll
  for (int a = 0; a < 4; ++a)
    #pragma unroll
    for (int b = 0; b < 5; ++b)
      #pragma unroll
      for (int r = 0; r < 8; ++r) acc[a][b][r] = 0.0f;

  // ---------------- Layer 1: h1 = gelu(x @ w1^T), K = 1536 ----------------
  // Prologue: DMA chunk 0 into f32 stage buffer 0.
  stage_chunk_async(x, m0, 0, lds + OFF_XF32, tid);
  asm volatile("s_wait_asynccnt 0x0" ::: "memory");
  __syncthreads();

  const _Float16* aBase1 = xs + l16 * KC + hsel * 8;
  const _Float16* bBase1 = w1 + (size_t)(wave * 80 + l16) * IN_F + hsel * 8;

  for (int c = 0; c < NCHUNK; ++c) {
    const int kc = c * KC;
    // Kick off DMA for the next chunk into the other buffer; overlaps compute.
    if (c + 1 < NCHUNK)
      stage_chunk_async(x, m0, kc + KC, lds + OFF_XF32 + ((c + 1) & 1) * XF32_BYTES, tid);

    // Convert current f32 chunk -> f16 xs (identical linear layout, f*4 halves).
    const float* fb = (const float*)(lds + OFF_XF32 + (c & 1) * XF32_BYTES);
    #pragma unroll
    for (int i = 0; i < 8; ++i) {
      int f = tid + i * 256;
      float4 v = *(const float4*)(fb + (size_t)f * 4);
      v4h hv;
      hv[0] = (_Float16)v.x; hv[1] = (_Float16)v.y;
      hv[2] = (_Float16)v.z; hv[3] = (_Float16)v.w;
      *(v4h*)(xs + (size_t)f * 4) = hv;
    }
    __syncthreads();   // xs ready for all waves

    gemm_block<KC / 32, KC, IN_F>(aBase1, bBase1 + kc, acc);

    // Next chunk's DMA must have landed, and xs consumption must finish
    // before it is overwritten next iteration.
    asm volatile("s_wait_asynccnt 0x0" ::: "memory");
    __syncthreads();
  }

  #pragma unroll
  for (int ms = 0; ms < 4; ++ms)
    #pragma unroll
    for (int t = 0; t < 5; ++t) {
      int n0 = (wave * 5 + t) * 16;
      #pragma unroll
      for (int r = 0; r < 8; ++r) {
        int row = ms * 16 + r + hsel * 8;
        h1[row * HID + n0 + l16] = (_Float16)gelu_exact(acc[ms][t][r]);
      }
    }
  __syncthreads();

  // ---------------- Layer 2: h2 = gelu(h1 @ w2^T), K = 640 ----------------
  #pragma unroll
  for (int a = 0; a < 4; ++a)
    #pragma unroll
    for (int b = 0; b < 5; ++b)
      #pragma unroll
      for (int r = 0; r < 8; ++r) acc[a][b][r] = 0.0f;

  gemm_block<HID / 32, HID, HID>(h1 + l16 * HID + hsel * 8,
                                 w2 + (size_t)(wave * 80 + l16) * HID + hsel * 8,
                                 acc);

  #pragma unroll
  for (int ms = 0; ms < 4; ++ms)
    #pragma unroll
    for (int t = 0; t < 5; ++t) {
      int n0 = (wave * 5 + t) * 16;
      #pragma unroll
      for (int r = 0; r < 8; ++r) {
        int row = ms * 16 + r + hsel * 8;
        h2[row * HID + n0 + l16] = (_Float16)gelu_exact(acc[ms][t][r]);
      }
    }
  __syncthreads();

  // ---------------- Layer 3: out = h2 @ w3^T (N padded 10->16) ------------
  if (wave < 4) {                 // wave-uniform: EXEC all-ones inside
    const int ms = wave;
    v8f acc3;
    #pragma unroll
    for (int r = 0; r < 8; ++r) acc3[r] = 0.0f;
    for (int k0 = 0; k0 < HID; k0 += 32) {
      const _Float16* ab = h2 + (ms * 16 + l16) * HID + k0 + hsel * 8;
      HFrag afr;
      afr.h[0] = *(const v8h*)(ab);
      afr.h[1] = *(const v8h*)(ab + 16);
      const _Float16* bb = w3 + l16 * HID + k0 + hsel * 8;
      HFrag bfr;
      bfr.h[0] = *(const v8h*)(bb);
      bfr.h[1] = *(const v8h*)(bb + 16);
      acc3 = __builtin_amdgcn_wmma_f32_16x16x32_f16(
          false, afr.v, false, bfr.v, (short)0, acc3, false, false);
    }
    if (l16 < OUT_F) {
      #pragma unroll
      for (int r = 0; r < 8; ++r) {
        int row = m0 + ms * 16 + r + hsel * 8;
        out[(size_t)row * OUT_F + l16] = acc3[r];
      }
    }
  }
}

extern "C" void kernel_launch(void* const* d_in, const int* in_sizes, int n_in,
                              void* d_out, int out_size, void* d_ws, size_t ws_size,
                              hipStream_t stream) {
  (void)in_sizes; (void)n_in; (void)out_size; (void)ws_size;

  const float* x  = (const float*)d_in[0];
  const float* W1 = (const float*)d_in[1];
  const float* W2 = (const float*)d_in[2];
  const float* W3 = (const float*)d_in[3];
  const unsigned char* m1 = (const unsigned char*)d_in[4];
  const unsigned char* m2 = (const unsigned char*)d_in[5];
  const unsigned char* m3 = (const unsigned char*)d_in[6];
  float* out = (float*)d_out;

  _Float16* w1h = (_Float16*)d_ws;
  _Float16* w2h = (_Float16*)((char*)d_ws + W1_BYTES);
  _Float16* w3h = (_Float16*)((char*)d_ws + W1_BYTES + W2_BYTES);

  sparse_mlp_prep<<<(HID * IN_F + 255) / 256, 256, 0, stream>>>(W1, m1, w1h, IN_F, HID, HID);
  sparse_mlp_prep<<<(HID * HID + 255) / 256, 256, 0, stream>>>(W2, m2, w2h, HID, HID, HID);
  sparse_mlp_prep<<<(W3_ROWS * HID + 255) / 256, 256, 0, stream>>>(W3, m3, w3h, HID, W3_ROWS, OUT_F);

  (void)hipFuncSetAttribute(reinterpret_cast<const void*>(sparse_mlp_fused),
                            hipFuncAttributeMaxDynamicSharedMemorySize, (int)LDS_BYTES);
  sparse_mlp_fused<<<BATCH / MTILE, 256, LDS_BYTES, stream>>>(x, w1h, w2h, w3h, out);
}
